// Gemma2Attention_59193239273825
// MI455X (gfx1250) — compile-verified, matrix operand order checked by reading, not run on hardware
//
#include <hip/hip_runtime.h>
#include <hip/hip_bf16.h>

// ---------------- problem constants ----------------
#define HIDd 3584
#define NHd  16
#define NKVd 8
#define HDd  256
#define Td   8192
#define Bd   4
#define Sd   2048
#define QD   (NHd * HDd)   // 4096
#define KVD  (NKVd * HDd)  // 2048
#define SOFTCAPc 50.0f
#define SCALEc   0.0625f   // 256^-0.5

typedef __attribute__((ext_vector_type(16))) __bf16 v16bf;
typedef __attribute__((ext_vector_type(8)))  float  v8f;

union FragU { v16bf v; unsigned u[8]; };

__device__ __forceinline__ unsigned short f2bf(float f) {
    union { float f; unsigned u; } v; v.f = f;
    unsigned r = v.u + 0x7fffu + ((v.u >> 16) & 1u);
    return (unsigned short)(r >> 16);
}
__device__ __forceinline__ float bf2f(unsigned short b) {
    union { unsigned u; float f; } v; v.u = ((unsigned)b) << 16;
    return v.f;
}

// Generic->LDS offset: low 32 bits of a generic pointer into LDS are the
// LDS byte offset (aperture handled by hardware).
__device__ __forceinline__ unsigned lds_off_of(const void* p) {
    return (unsigned)(unsigned long long)p;
}

// CDNA5 async copy: global -> LDS, 16B per lane, tracked by ASYNCcnt.
__device__ __forceinline__ void async_load_b128(unsigned lds_off, const void* gptr) {
    asm volatile("global_load_async_to_lds_b128 %0, %1, off"
                 :: "v"(lds_off), "v"(gptr) : "memory");
}
__device__ __forceinline__ void wait_async0() {
    asm volatile("s_wait_asynccnt 0" ::: "memory");
}

// Load a 16x32 bf16 A-style fragment (ISA 7.12.2 layout) from row-major
// storage with row pitch `ld` (in elements). Works for A tiles and for
// B tiles stored column(N)-major (so both use contiguous K reads).
__device__ __forceinline__ v16bf lds_load_frag(const unsigned short* base, int ld, int lane) {
    const int m = lane & 15;
    const int h = lane >> 4;
    FragU f;
    const unsigned short* rp = base + m * ld;
#pragma unroll
    for (int j = 0; j < 8; ++j) {
        int k = ((j < 4) ? (2 * j) : (16 + 2 * (j - 4))) + 8 * h;
        f.u[j] = *(const unsigned*)(rp + k);
    }
    return f.v;
}

// ---------------- fp32 -> bf16 convert ----------------
__global__ void f32_to_bf16_kernel(const float* __restrict__ src,
                                   unsigned short* __restrict__ dst, long long n) {
    for (long long i = blockIdx.x * (long long)blockDim.x + threadIdx.x; i < n;
         i += (long long)gridDim.x * blockDim.x)
        dst[i] = f2bf(src[i]);
}

// ---------------- tiled bf16 GEMM (C = A @ B) ----------------
// A: MxK row-major bf16, B: KxN row-major bf16, C: MxN (f32 or bf16)
#define GBM 128
#define GBN 64
#define GBK 32
#define GLD 40  // 80B row pitch: 16B-aligned rows for async b128, low-conflict frags

__global__ __launch_bounds__(256) void gemm_bf16_kernel(
    const unsigned short* __restrict__ A, const unsigned short* __restrict__ B,
    float* __restrict__ Cf, unsigned short* __restrict__ Cb, int M, int N, int K) {
    __shared__ unsigned short As[GBM][GLD];
    __shared__ unsigned short Bst[GBN][GLD];  // B tile stored N-major (transposed)

    const int tid  = threadIdx.x;
    const int lane = tid & 31;
    const int wave = tid >> 5;
    const int wm   = (wave & 3) * 32;
    const int wn   = (wave >> 2) * 32;
    const int m0   = blockIdx.y * GBM;
    const int n0   = blockIdx.x * GBN;

    v8f acc[2][2];
#pragma unroll
    for (int i = 0; i < 2; ++i)
#pragma unroll
        for (int j = 0; j < 2; ++j) acc[i][j] = {};

    const int nk = K / GBK;
    for (int kt = 0; kt < nk; ++kt) {
        const int k0 = kt * GBK;
        // A tile: 128x32 bf16 = 512 x b128, async global->LDS (2 per thread)
#pragma unroll
        for (int i = 0; i < 2; ++i) {
            int d = tid + 256 * i;
            int r = d >> 2;
            int c = (d & 3) * 8;
            async_load_b128(lds_off_of(&As[r][c]),
                            &A[(size_t)(m0 + r) * K + k0 + c]);
        }
        // B tile: 32x64 bf16, gather 4 dwords into regs then transpose-store
        unsigned bv[4];
#pragma unroll
        for (int i = 0; i < 4; ++i) {
            int d = tid + 256 * i;
            int r = d >> 5;
            int c = (d & 31) * 2;
            bv[i] = *(const unsigned*)&B[(size_t)(k0 + r) * N + n0 + c];
        }
#pragma unroll
        for (int i = 0; i < 4; ++i) {
            int d = tid + 256 * i;
            int r = d >> 5;
            int c = (d & 31) * 2;
            Bst[c][r]     = (unsigned short)(bv[i] & 0xffffu);
            Bst[c + 1][r] = (unsigned short)(bv[i] >> 16);
        }
        if (kt + 1 < nk) {  // global_prefetch_b8 of next A tile
            __builtin_prefetch(&A[(size_t)(m0 + (tid & 127)) * K + k0 + GBK], 0, 0);
        }
        wait_async0();
        __syncthreads();

        v16bf af[2], bfr[2];
#pragma unroll
        for (int i = 0; i < 2; ++i) af[i] = lds_load_frag(&As[wm + 16 * i][0], GLD, lane);
#pragma unroll
        for (int j = 0; j < 2; ++j) bfr[j] = lds_load_frag(&Bst[wn + 16 * j][0], GLD, lane);
#pragma unroll
        for (int i = 0; i < 2; ++i)
#pragma unroll
            for (int j = 0; j < 2; ++j)
                acc[i][j] = __builtin_amdgcn_wmma_f32_16x16x32_bf16(
                    false, af[i], false, bfr[j], (short)0, acc[i][j], false, false);
        __syncthreads();
    }

    const int h  = lane >> 4;
    const int nl = lane & 15;
#pragma unroll
    for (int i = 0; i < 2; ++i)
#pragma unroll
        for (int j = 0; j < 2; ++j)
#pragma unroll
            for (int r = 0; r < 8; ++r) {
                int row = m0 + wm + 16 * i + r + 8 * h;
                int col = n0 + wn + 16 * j + nl;
                float v = acc[i][j][r];
                if (Cf) Cf[(size_t)row * N + col] = v;
                else    Cb[(size_t)row * N + col] = f2bf(v);
            }
}

// ---------------- RoPE (rotate-half) in-place on bf16 Q/K ----------------
__global__ void rope_kernel(unsigned short* __restrict__ q, unsigned short* __restrict__ k,
                            const float* __restrict__ inv_freq, const int* __restrict__ pos) {
    const long long total = (long long)Td * (NHd + NKVd) * (HDd / 2);
    for (long long idx = blockIdx.x * (long long)blockDim.x + threadIdx.x; idx < total;
         idx += (long long)gridDim.x * blockDim.x) {
        int i  = (int)(idx % (HDd / 2));
        long long r = idx / (HDd / 2);
        int hh = (int)(r % (NHd + NKVd));
        int t  = (int)(r / (NHd + NKVd));
        unsigned short* p;
        if (hh < NHd) p = q + (size_t)t * QD + hh * HDd;
        else          p = k + (size_t)t * KVD + (hh - NHd) * HDd;
        float f = (float)pos[t] * inv_freq[i];
        float c = __cosf(f), s = __sinf(f);
        float x1 = bf2f(p[i]);
        float x2 = bf2f(p[i + HDd / 2]);
        p[i]           = f2bf(x1 * c - x2 * s);
        p[i + HDd / 2] = f2bf(x2 * c + x1 * s);
    }
}

// ---------------- flash attention (bf16 WMMA, online softmax, softcap) ----------------
#define QT  64   // q rows per block (4 waves x 16)
#define KT  32   // key rows per iteration
#define KLD 264  // 528B row pitch: 16B-aligned rows for async b128
#define VLD 34
#define PLD 34

__global__ __launch_bounds__(128) void attn_kernel(
    const unsigned short* __restrict__ Q, const unsigned short* __restrict__ Kc,
    const unsigned short* __restrict__ Vc, unsigned short* __restrict__ Oo) {
    __shared__ unsigned short Ks[KT][KLD];     // [key][hd]
    __shared__ unsigned short Vst[HDd][VLD];   // transposed: [hd][key]
    __shared__ unsigned short Ps[4][16][PLD];  // per-wave P staging [qrow][key]

    const int tid   = threadIdx.x;
    const int lane  = tid & 31;
    const int wave  = tid >> 5;
    const int h     = lane >> 4;
    const int nl    = lane & 15;
    const int qb    = blockIdx.x * QT;
    const int head  = blockIdx.y;
    const int batch = blockIdx.z;
    const int kvh   = head >> 1;  // GQA: NH/NKV = 2

    // Q fragments for this wave's 16 rows, kept in registers (8 x v16bf)
    v16bf qf[8];
    {
        const size_t qrow = (size_t)(batch * Sd + qb + wave * 16 + nl);
        const unsigned short* qp = Q + qrow * QD + head * HDd;
#pragma unroll
        for (int c = 0; c < 8; ++c) {
            FragU f;
#pragma unroll
            for (int j = 0; j < 8; ++j) {
                int k = ((j < 4) ? (2 * j) : (16 + 2 * (j - 4))) + 8 * h;
                f.u[j] = *(const unsigned*)(qp + c * 32 + k);
            }
            qf[c] = f.v;
        }
    }

    v8f Oacc[16];
#pragma unroll
    for (int c = 0; c < 16; ++c) Oacc[c] = {};
    float m_i[8], l_i[8];
#pragma unroll
    for (int r = 0; r < 8; ++r) { m_i[r] = -1e30f; l_i[r] = 0.0f; }

    const int nkt = qb / KT + 2;  // causal: keys up to qb+63
    for (int kt = 0; kt < nkt; ++kt) {
        __syncthreads();
        // K tile 32x256: 1024 x b128, async global->LDS (8 per thread)
#pragma unroll
        for (int i = 0; i < 8; ++i) {
            int d = tid + 128 * i;
            int r = d >> 5;
            int c = (d & 31) * 8;
            async_load_b128(lds_off_of(&Ks[r][c]),
                            &Kc[((size_t)(batch * Sd + kt * KT + r)) * KVD + kvh * HDd + c]);
        }
        // V tile transposed into Vst[hd][key]: 4 passes of 8-dword gathers
#pragma unroll
        for (int pass = 0; pass < 4; ++pass) {
            unsigned vv[8];
#pragma unroll
            for (int i = 0; i < 8; ++i) {
                int d = tid + 128 * (pass * 8 + i);
                int r = d >> 7;
                int c = (d & 127) * 2;
                vv[i] = *(const unsigned*)&Vc[((size_t)(batch * Sd + kt * KT + r)) * KVD +
                                              kvh * HDd + c];
            }
#pragma unroll
            for (int i = 0; i < 8; ++i) {
                int d = tid + 128 * (pass * 8 + i);
                int r = d >> 7;
                int c = (d & 127) * 2;
                Vst[c][r]     = (unsigned short)(vv[i] & 0xffffu);
                Vst[c + 1][r] = (unsigned short)(vv[i] >> 16);
            }
        }
        wait_async0();
        __syncthreads();

        // scores: S = Q . K^T over HD=256 (8 wmma k-steps), 2 key sub-tiles
        v8f sacc[2];
#pragma unroll
        for (int n2 = 0; n2 < 2; ++n2) {
            sacc[n2] = {};
#pragma unroll
            for (int c = 0; c < 8; ++c) {
                v16bf bk = lds_load_frag(&Ks[n2 * 16][c * 32], KLD, lane);
                sacc[n2] = __builtin_amdgcn_wmma_f32_16x16x32_bf16(
                    false, qf[c], false, bk, (short)0, sacc[n2], false, false);
            }
        }
        // scale + softcap tanh + causal mask
        float sv[2][8];
#pragma unroll
        for (int n2 = 0; n2 < 2; ++n2)
#pragma unroll
            for (int r = 0; r < 8; ++r) {
                float s = sacc[n2][r] * SCALEc;
                s = SOFTCAPc * tanhf(s * (1.0f / SOFTCAPc));
                int qrow = qb + wave * 16 + r + 8 * h;
                int kcol = kt * KT + n2 * 16 + nl;
                sv[n2][r] = (kcol > qrow) ? -1e30f : s;
            }
        // online softmax per row (row reduction across 16-lane half via shfl_xor)
#pragma unroll
        for (int r = 0; r < 8; ++r) {
            float rm = fmaxf(sv[0][r], sv[1][r]);
            rm = fmaxf(rm, __shfl_xor(rm, 1, 32));
            rm = fmaxf(rm, __shfl_xor(rm, 2, 32));
            rm = fmaxf(rm, __shfl_xor(rm, 4, 32));
            rm = fmaxf(rm, __shfl_xor(rm, 8, 32));
            float mn    = fmaxf(m_i[r], rm);
            float alpha = __expf(m_i[r] - mn);
            float p0    = __expf(sv[0][r] - mn);
            float p1    = __expf(sv[1][r] - mn);
            float rs    = p0 + p1;
            rs += __shfl_xor(rs, 1, 32);
            rs += __shfl_xor(rs, 2, 32);
            rs += __shfl_xor(rs, 4, 32);
            rs += __shfl_xor(rs, 8, 32);
            l_i[r] = l_i[r] * alpha + rs;
            m_i[r] = mn;
#pragma unroll
            for (int c = 0; c < 16; ++c) Oacc[c][r] *= alpha;
            // stage P (C-layout -> A-layout via LDS), wave-private region
            Ps[wave][r + 8 * h][nl]      = f2bf(p0);
            Ps[wave][r + 8 * h][16 + nl] = f2bf(p1);
        }
        // O += P . V  (contract 32 keys, 16 hd chunks)
        v16bf pf = lds_load_frag(&Ps[wave][0][0], PLD, lane);
#pragma unroll
        for (int c = 0; c < 16; ++c) {
            v16bf vf = lds_load_frag(&Vst[c * 16][0], VLD, lane);
            Oacc[c] = __builtin_amdgcn_wmma_f32_16x16x32_bf16(
                false, pf, false, vf, (short)0, Oacc[c], false, false);
        }
    }

    // normalize and store bf16 attn output [T][NH*HD]
#pragma unroll
    for (int r = 0; r < 8; ++r) {
        float inv = 1.0f / l_i[r];
#pragma unroll
        for (int c = 0; c < 16; ++c) {
            int row = batch * Sd + qb + wave * 16 + r + 8 * h;
            int col = head * HDd + c * 16 + nl;
            Oo[(size_t)row * QD + col] = f2bf(Oacc[c][r] * inv);
        }
    }
}

// ---------------- launch ----------------
extern "C" void kernel_launch(void* const* d_in, const int* in_sizes, int n_in,
                              void* d_out, int out_size, void* d_ws, size_t ws_size,
                              hipStream_t stream) {
    (void)in_sizes; (void)n_in; (void)out_size; (void)ws_size;
    const float* X        = (const float*)d_in[0];
    const float* Wq       = (const float*)d_in[1];
    const float* Wk       = (const float*)d_in[2];
    const float* Wv       = (const float*)d_in[3];
    const float* Wo       = (const float*)d_in[4];
    const float* inv_freq = (const float*)d_in[5];
    const int*   pos      = (const int*)d_in[6];
    // d_in[7] = cu_seqlens (uniform, encoded via Bd/Sd constants)

    char* ws = (char*)d_ws;
    size_t off = 0;
    auto alloc = [&](size_t bytes) {
        void* p = ws + off;
        off += (bytes + 255) & ~(size_t)255;
        return p;
    };
    unsigned short* Xb  = (unsigned short*)alloc((size_t)Td * HIDd * 2);
    unsigned short* Wqb = (unsigned short*)alloc((size_t)HIDd * QD * 2);
    unsigned short* Wkb = (unsigned short*)alloc((size_t)HIDd * KVD * 2);
    unsigned short* Wvb = (unsigned short*)alloc((size_t)HIDd * KVD * 2);
    unsigned short* Wob = (unsigned short*)alloc((size_t)QD * HIDd * 2);
    unsigned short* Qb  = (unsigned short*)alloc((size_t)Td * QD * 2);
    unsigned short* Kb  = (unsigned short*)alloc((size_t)Td * KVD * 2);
    unsigned short* Vb  = (unsigned short*)alloc((size_t)Td * KVD * 2);
    unsigned short* Ab  = (unsigned short*)alloc((size_t)Td * QD * 2);

    f32_to_bf16_kernel<<<4096, 256, 0, stream>>>(X, Xb, (long long)Td * HIDd);
    f32_to_bf16_kernel<<<4096, 256, 0, stream>>>(Wq, Wqb, (long long)HIDd * QD);
    f32_to_bf16_kernel<<<4096, 256, 0, stream>>>(Wk, Wkb, (long long)HIDd * KVD);
    f32_to_bf16_kernel<<<4096, 256, 0, stream>>>(Wv, Wvb, (long long)HIDd * KVD);
    f32_to_bf16_kernel<<<4096, 256, 0, stream>>>(Wo, Wob, (long long)QD * HIDd);

    gemm_bf16_kernel<<<dim3(QD / GBN, Td / GBM), 256, 0, stream>>>(
        Xb, Wqb, nullptr, Qb, Td, QD, HIDd);
    gemm_bf16_kernel<<<dim3(KVD / GBN, Td / GBM), 256, 0, stream>>>(
        Xb, Wkb, nullptr, Kb, Td, KVD, HIDd);
    gemm_bf16_kernel<<<dim3(KVD / GBN, Td / GBM), 256, 0, stream>>>(
        Xb, Wvb, nullptr, Vb, Td, KVD, HIDd);

    rope_kernel<<<4096, 256, 0, stream>>>(Qb, Kb, inv_freq, pos);

    attn_kernel<<<dim3(Sd / QT, NHd, Bd), 128, 0, stream>>>(Qb, Kb, Vb, Ab);

    gemm_bf16_kernel<<<dim3(HIDd / GBN, Td / GBM), 256, 0, stream>>>(
        Ab, Wob, (float*)d_out, nullptr, Td, HIDd, QD);
}
